// GraphMAE_2800318677028
// MI455X (gfx1250) — matrix operand order repaired and checked by reading, not compiled, not run on hardware
//
#include <hip/hip_runtime.h>
#include <hip/hip_bf16.h>

typedef __bf16 bf16_t;
typedef __attribute__((ext_vector_type(16))) __bf16 v16bf;
typedef __attribute__((ext_vector_type(8)))  float  v8f;

#define NEG_SLOPE 0.2f

// ---------------------------------------------------------------------------
// utility kernels
// ---------------------------------------------------------------------------
__global__ void zero_f32(float* __restrict__ p, long long n) {
    long long t = (long long)blockIdx.x * blockDim.x + threadIdx.x;
    if (t < n) p[t] = 0.0f;
}

__global__ void fill_neginf(float* __restrict__ p, int n) {
    int t = blockIdx.x * blockDim.x + threadIdx.x;
    if (t < n) p[t] = __int_as_float(0xFF800000); // -inf
}

// x = arange(N) in the reference, but stay faithful: gather emb_table[x]
__global__ void gather_rows(const float* __restrict__ emb, const int* __restrict__ x,
                            float* __restrict__ out, int N, int H) {
    long long t = (long long)blockIdx.x * blockDim.x + threadIdx.x;
    if (t >= (long long)N * H) return;
    int i = (int)(t / H), j = (int)(t % H);
    out[t] = emb[(long long)x[i] * H + j];
}

__global__ void set_rows(float* __restrict__ X, const int* __restrict__ sel,
                         const float* __restrict__ tok, int M, int H) {
    int t = blockIdx.x * blockDim.x + threadIdx.x;
    if (t >= M * H) return;
    int m = t / H, j = t % H;
    X[(long long)sel[m] * H + j] = tok[j];
}

// ---------------------------------------------------------------------------
// C[N,H] = A[N,H] @ W[H,H]^T  via v_wmma_f32_16x16x32_bf16.
// One wave computes one 16x16 C tile; K loop = H/32 WMMA steps.
// VGPR layouts per CDNA5 ISA §7.12.2:
//   A (16x32 bf16): lane<16 -> M=lane,   K = {v<4: 2v+h,  v>=4: 16+2(v-4)+h}
//                   lane>=16 -> M=lane-16, same K pattern +8
//   B (32x16 bf16): lane gives N=lane%16; halves give K = (lane>=16?8:0)+e, +16 for e>=8 (as A)
//   C (8xf32):      VGPR r: M = r + (lane>=16?8:0), N = lane%16
// EXEC is all-1 per wave (guard is wave-uniform).
// ---------------------------------------------------------------------------
__global__ void gemm_bt_wmma(const float* __restrict__ A, const float* __restrict__ W,
                             float* __restrict__ C, int N, int H) {
    const int lane  = threadIdx.x & 31;
    const int wave  = (int)(((long long)blockIdx.x * blockDim.x + threadIdx.x) >> 5);
    const int tilesN = H >> 4;                 // 8 for H=128
    const int tiles  = (N >> 4) * tilesN;
    if (wave >= tiles) return;
    const int tm = wave / tilesN;
    const int tn = wave % tilesN;
    const int l15 = lane & 15;
    const int hiA = (lane >= 16) ? 8  : 0;     // A K-split per half-wave
    const int hiB = (lane >= 16) ? 16 : 0;     // unused; B uses hiA (same K split as A)

    const float* __restrict__ arow = A + (long long)(tm * 16 + l15) * H;
    const float* __restrict__ wrow = W + (long long)(tn * 16 + l15) * H;

    v8f acc = {};
#pragma unroll
    for (int kb = 0; kb < 128; kb += 32) {
        const float* ap = arow + kb;
        const float* wp = wrow + kb;
        v16bf a, b;
#pragma unroll
        for (int e = 0; e < 8; ++e) {
            a[e]     = (bf16_t)ap[hiA + e];        // v=0..3 -> K in [hiA, hiA+7]
            a[e + 8] = (bf16_t)ap[16 + hiA + e];   // v=4..7 -> K in [16+hiA, ...]
            b[e]     = (bf16_t)wp[hiA + e];        // K same as A
            b[e + 8] = (bf16_t)wp[16 + hiA + e];
        }
        acc = __builtin_amdgcn_wmma_f32_16x16x32_bf16(
            false, a, false, b, (short)0, acc, false, false);
    }

    const int colC = tn * 16 + l15;
    float* __restrict__ crow = C + (long long)(tm * 16 + hiA) * H + colC;
#pragma unroll
    for (int r = 0; r < 8; ++r)
        crow[(long long)r * H] = acc[r];
}

// ---------------------------------------------------------------------------
// a_src[i] = dot(x_p[i,:], att_src) ; a_dst[i] = dot(x_p[i,:], att_dst)
// one wave per node, wave32 butterfly reduction
// ---------------------------------------------------------------------------
__global__ void att_gemv(const float* __restrict__ xp, const float* __restrict__ as,
                         const float* __restrict__ ad, float* __restrict__ out_s,
                         float* __restrict__ out_d, int N, int H) {
    const int wave = (int)(((long long)blockIdx.x * blockDim.x + threadIdx.x) >> 5);
    const int lane = threadIdx.x & 31;
    if (wave >= N) return;
    float ss = 0.f, sd = 0.f;
    const float* row = xp + (long long)wave * H;
    for (int j = lane; j < H; j += 32) {
        float v = row[j];
        ss += v * as[j];
        sd += v * ad[j];
    }
#pragma unroll
    for (int off = 16; off > 0; off >>= 1) {
        ss += __shfl_xor(ss, off, 32);
        sd += __shfl_xor(sd, off, 32);
    }
    if (lane == 0) { out_s[wave] = ss; out_d[wave] = sd; }
}

// ---------------------------------------------------------------------------
// edge pass 1: e = leaky_relu(a_src[src] + a_dst[dst]); segment max over dst
// float atomic-max via sign-aware int/uint atomics (init = -inf bits)
// ---------------------------------------------------------------------------
__device__ __forceinline__ void atomicMaxF32(float* addr, float v) {
    if (v >= 0.0f) atomicMax((int*)addr, __float_as_int(v));
    else           atomicMin((unsigned int*)addr, __float_as_uint(v));
}

__global__ void edge_max_k(const int* __restrict__ ei, int E, int N,
                           const float* __restrict__ a_s, const float* __restrict__ a_d,
                           float* __restrict__ e_buf, float* __restrict__ seg_max) {
    int e = blockIdx.x * blockDim.x + threadIdx.x;
    int ET = E + N;
    if (e >= ET) return;
    int s, d;
    if (e < E) { s = ei[e]; d = ei[E + e]; } else { s = d = e - E; }
    float v = a_s[s] + a_d[d];
    v = (v > 0.0f) ? v : NEG_SLOPE * v;
    e_buf[e] = v;
    atomicMaxF32(seg_max + d, v);
}

// ---------------------------------------------------------------------------
// edge pass 2: w = exp(e - max[dst]); denom[dst]+=w; agg[dst,:] += w*x_p[src,:]
// 128 threads per edge (one per feature)
// ---------------------------------------------------------------------------
__global__ void edge_accum_k(const int* __restrict__ ei, int E, int N, int H,
                             const float* __restrict__ e_buf, const float* __restrict__ seg_max,
                             const float* __restrict__ xp, float* __restrict__ denom,
                             float* __restrict__ agg) {
    long long t = (long long)blockIdx.x * blockDim.x + threadIdx.x;
    long long ET = (long long)E + N;
    long long e = t >> 7;          // H == 128
    int j = (int)(t & 127);
    if (e >= ET) return;
    int s, d;
    if (e < E) { s = ei[e]; d = ei[E + e]; } else { s = d = (int)(e - E); }
    float w = __expf(e_buf[e] - seg_max[d]);
    if (j == 0) atomicAdd(denom + d, w);
    atomicAdd(agg + (long long)d * H + j, w * xp[(long long)s * H + j]);
}

// out[i,j] = agg[i,j]/denom[i] + bias[j]   (self-loops guarantee denom > 0)
__global__ void finalize_k(float* __restrict__ agg, const float* __restrict__ denom,
                           const float* __restrict__ bias, int N, int H) {
    long long t = (long long)blockIdx.x * blockDim.x + threadIdx.x;
    if (t >= (long long)N * H) return;
    int i = (int)(t / H), j = (int)(t % H);
    agg[t] = agg[t] / denom[i] + bias[j];
}

// ---------------------------------------------------------------------------
// loss: origin rows == mask_tok (reference sets them before encoding), so
// xn is normalize(mask_tok); per masked row compute (1 - cos(xn, x_dec))^3
// ---------------------------------------------------------------------------
__global__ void loss_k(const float* __restrict__ xdec, const int* __restrict__ sel,
                       const float* __restrict__ tok, float* __restrict__ acc,
                       int M, int H) {
    const int wave = (int)(((long long)blockIdx.x * blockDim.x + threadIdx.x) >> 5);
    const int lane = threadIdx.x & 31;
    if (wave >= M) return;
    const float* row = xdec + (long long)sel[wave] * H;
    float nx = 0.f, ny = 0.f, dp = 0.f;
    for (int j = lane; j < H; j += 32) {
        float a = tok[j], b = row[j];
        nx += a * a; ny += b * b; dp += a * b;
    }
#pragma unroll
    for (int off = 16; off > 0; off >>= 1) {
        nx += __shfl_xor(nx, off, 32);
        ny += __shfl_xor(ny, off, 32);
        dp += __shfl_xor(dp, off, 32);
    }
    if (lane == 0) {
        float c = dp / (fmaxf(sqrtf(nx), 1e-12f) * fmaxf(sqrtf(ny), 1e-12f));
        float l = 1.0f - c;
        atomicAdd(acc, l * l * l);
    }
}

__global__ void final_scalar_k(const float* __restrict__ acc, float* __restrict__ out,
                               float invM) {
    if (threadIdx.x == 0 && blockIdx.x == 0) out[0] = acc[0] * invM;
}

// ---------------------------------------------------------------------------
extern "C" void kernel_launch(void* const* d_in, const int* in_sizes, int n_in,
                              void* d_out, int out_size, void* d_ws, size_t ws_size,
                              hipStream_t stream) {
    const float* emb      = (const float*)d_in[0];
    const float* W_enc    = (const float*)d_in[1];
    const float* as_enc   = (const float*)d_in[2];
    const float* ad_enc   = (const float*)d_in[3];
    const float* b_enc    = (const float*)d_in[4];
    const float* W_e2d    = (const float*)d_in[5];
    const float* W_dec    = (const float*)d_in[6];
    const float* as_dec   = (const float*)d_in[7];
    const float* ad_dec   = (const float*)d_in[8];
    const float* b_dec    = (const float*)d_in[9];
    const float* mask_tok = (const float*)d_in[10];
    const float* dmask_tok= (const float*)d_in[11];
    const int*   x        = (const int*)d_in[12];
    const int*   ei       = (const int*)d_in[13];
    const int*   sel      = (const int*)d_in[14];

    const int N = in_sizes[12];
    const int H = in_sizes[10];          // 128
    const int E = in_sizes[13] / 2;
    const int M = in_sizes[14];
    const int ET = E + N;
    const long long NH = (long long)N * H;

    // workspace layout (floats)
    float* buf0  = (float*)d_ws;         // x_masked, later x_rep
    float* buf1  = buf0 + NH;            // x_p
    float* buf2  = buf1 + NH;            // agg -> x_enc / x_dec
    float* a_s   = buf2 + NH;
    float* a_d   = a_s + N;
    float* smax  = a_d + N;
    float* den   = smax + N;
    float* ebuf  = den + N;              // ET floats
    float* acc   = ebuf + ET;            // 1 float

    const int B = 256;
    auto cdiv = [](long long a, long long b) { return (int)((a + b - 1) / b); };

    const int tiles     = (N / 16) * (H / 16);
    const int gemm_blk  = cdiv((long long)tiles * 32, B);
    const int nh_blk    = cdiv(NH, B);
    const int node_blk  = cdiv((long long)N * 32, B);
    const int edge_blk  = cdiv(ET, B);
    const int eacc_blk  = cdiv((long long)ET * H, B);
    const int mh_blk    = cdiv((long long)M * H, B);
    const int loss_blk  = cdiv((long long)M * 32, B);

    // ---- encoder ----
    gather_rows<<<nh_blk, B, 0, stream>>>(emb, x, buf0, N, H);
    set_rows<<<mh_blk, B, 0, stream>>>(buf0, sel, mask_tok, M, H);
    gemm_bt_wmma<<<gemm_blk, B, 0, stream>>>(buf0, W_enc, buf1, N, H);
    att_gemv<<<node_blk, B, 0, stream>>>(buf1, as_enc, ad_enc, a_s, a_d, N, H);
    zero_f32<<<nh_blk, B, 0, stream>>>(buf2, NH);
    fill_neginf<<<cdiv(N, B), B, 0, stream>>>(smax, N);
    zero_f32<<<cdiv(N, B), B, 0, stream>>>(den, N);
    edge_max_k<<<edge_blk, B, 0, stream>>>(ei, E, N, a_s, a_d, ebuf, smax);
    edge_accum_k<<<eacc_blk, B, 0, stream>>>(ei, E, N, H, ebuf, smax, buf1, den, buf2);
    finalize_k<<<nh_blk, B, 0, stream>>>(buf2, den, b_enc, N, H);   // buf2 = x_enc

    // ---- e2d projection + decoder mask ----
    gemm_bt_wmma<<<gemm_blk, B, 0, stream>>>(buf2, W_e2d, buf0, N, H); // buf0 = x_rep
    set_rows<<<mh_blk, B, 0, stream>>>(buf0, sel, dmask_tok, M, H);

    // ---- decoder ----
    gemm_bt_wmma<<<gemm_blk, B, 0, stream>>>(buf0, W_dec, buf1, N, H);
    att_gemv<<<node_blk, B, 0, stream>>>(buf1, as_dec, ad_dec, a_s, a_d, N, H);
    zero_f32<<<nh_blk, B, 0, stream>>>(buf2, NH);
    fill_neginf<<<cdiv(N, B), B, 0, stream>>>(smax, N);
    zero_f32<<<cdiv(N, B), B, 0, stream>>>(den, N);
    edge_max_k<<<edge_blk, B, 0, stream>>>(ei, E, N, a_s, a_d, ebuf, smax);
    edge_accum_k<<<eacc_blk, B, 0, stream>>>(ei, E, N, H, ebuf, smax, buf1, den, buf2);
    finalize_k<<<nh_blk, B, 0, stream>>>(buf2, den, b_dec, N, H);   // buf2 = x_dec

    // ---- loss ----
    zero_f32<<<1, 32, 0, stream>>>(acc, 1);
    loss_k<<<loss_blk, B, 0, stream>>>(buf2, sel, mask_tok, acc, M, H);
    final_scalar_k<<<1, 32, 0, stream>>>(acc, (float*)d_out, 1.0f / (float)M);
}
